// FractalAttentionFusion_28312424415248
// MI455X (gfx1250) — matrix operand order repaired and checked by reading, not compiled, not run on hardware
//
#include <hip/hip_runtime.h>
#include <hip/hip_bf16.h>
#include <math.h>

// ---------------------------------------------------------------------------
// FractalAttentionFusion for MI455X (gfx1250, wave32, WMMA)
// D=768, B=16, PH=PW=32, N=1024, M_total=16384
// ---------------------------------------------------------------------------

typedef __attribute__((ext_vector_type(16))) _Float16 v16h;
typedef __attribute__((ext_vector_type(8)))  _Float16 v8h;
typedef __attribute__((ext_vector_type(8)))  float    v8f;

union Frag16 { v16h v; v8h h[2]; };

#define D_DIM 768
#define NSEQ  1024
#define NBATCH 16
#define MTOT  (NBATCH * NSEQ)

// ---- WMMA helpers ----------------------------------------------------------

__device__ __forceinline__ v8f wmma_f16(v16h a, v16h b, v8f c) {
  // (neg_a, A, neg_b, B, c_mod, C, reuse_a, reuse_b)
  return __builtin_amdgcn_wmma_f32_16x16x32_f16(false, a, false, b, (short)0, c,
                                                false, false);
}

// A-matrix fragment, 16x32 f16, row-major source (row = lane&15).
// Per ISA: lanes 0-15 VGPR0-3 hold K=0..7, VGPR4-7 hold K=16..23;
//          lanes 16-31 hold K=8..15 and K=24..31.
__device__ __forceinline__ v16h load_a(const _Float16* __restrict__ p, int ld, int lane) {
  int idx = lane & 15, hi = lane >> 4;
  const _Float16* base = p + (size_t)idx * ld + hi * 8;
  Frag16 f;
  f.h[0] = *(const v8h*)(base);        // K = hi*8 + 0..7
  f.h[1] = *(const v8h*)(base + 16);   // K = 16 + hi*8 + 0..7
  return f.v;
}

// Same A layout but converting from f32 source (used for LDS-resident P).
__device__ __forceinline__ v16h load_a_f32(const float* __restrict__ p, int ld, int lane) {
  int idx = lane & 15, hi = lane >> 4;
  const float* base = p + (size_t)idx * ld + hi * 8;
  Frag16 f;
#pragma unroll
  for (int e = 0; e < 8; ++e) {
    f.h[0][e] = (_Float16)base[e];
    f.h[1][e] = (_Float16)base[16 + e];
  }
  return f.v;
}

// B-matrix fragment, 32x16 f16, source row-major [N x K] (col = lane&15).
// Same K order as A: lanes 0-15 hold K=0..7 and K=16..23, lanes 16-31 K=8..15 and K=24..31.
__device__ __forceinline__ v16h load_b(const _Float16* __restrict__ p, int ld, int lane) {
  int idx = lane & 15, hi = lane >> 4;
  const _Float16* base = p + (size_t)idx * ld + hi * 8;
  Frag16 f;
  f.h[0] = *(const v8h*)(base);        // K = hi*8 + 0..7
  f.h[1] = *(const v8h*)(base + 16);   // K = 16 + hi*8 + 0..7
  return f.v;
}

// ---- wave32 reductions -----------------------------------------------------

__device__ __forceinline__ float wave_sum(float x) {
#pragma unroll
  for (int m = 16; m > 0; m >>= 1) x += __shfl_xor(x, m, 32);
  return x;
}
__device__ __forceinline__ float wave_max(float x) {
#pragma unroll
  for (int m = 16; m > 0; m >>= 1) x = fmaxf(x, __shfl_xor(x, m, 32));
  return x;
}

// ---------------------------------------------------------------------------
// Kernel: f32 -> f16 convert (feats, weights)
// ---------------------------------------------------------------------------
__global__ void cvt_f32_to_f16(const float* __restrict__ src,
                               _Float16* __restrict__ dst, int n) {
  int i = blockIdx.x * blockDim.x + threadIdx.x;
  if (i < n) dst[i] = (_Float16)src[i];
}

// ---------------------------------------------------------------------------
// Kernel: fractal scores -> standardized ns[b][n]. One block (256 thr) / batch.
// ---------------------------------------------------------------------------
__global__ __launch_bounds__(256)
void fractal_ns_kernel(const float* __restrict__ feats, float* __restrict__ ns) {
  __shared__ float energy[NSEQ];
  __shared__ float frac[NSEQ];
  __shared__ float rbuf[2][8];
  __shared__ float stat[2];

  const int b = blockIdx.x;
  const int t = threadIdx.x;
  const int lane = t & 31, wave = t >> 5;
  const float* fb = feats + (size_t)b * NSEQ * D_DIM;

  // energy[i] = mean_d feats[i,d]^2 ; one wave per row (coalesced over d)
  for (int i = wave; i < NSEQ; i += 8) {
    const float* row = fb + (size_t)i * D_DIM;
    float s = 0.f;
    for (int dd = lane; dd < D_DIM; dd += 32) { float x = row[dd]; s += x * x; }
    s = wave_sum(s);
    if (lane == 0) energy[i] = s * (1.0f / (float)D_DIM);
  }
  __syncthreads();

  // mean / std (ddof=1) of energy
  {
    float s1 = 0.f, s2 = 0.f;
    for (int i = t; i < NSEQ; i += 256) { float e = energy[i]; s1 += e; s2 += e * e; }
    s1 = wave_sum(s1); s2 = wave_sum(s2);
    if (lane == 0) { rbuf[0][wave] = s1; rbuf[1][wave] = s2; }
    __syncthreads();
    if (t == 0) {
      float a = 0.f, c = 0.f;
      for (int i = 0; i < 8; ++i) { a += rbuf[0][i]; c += rbuf[1][i]; }
      float mu = a / (float)NSEQ;
      float var = (c - (float)NSEQ * mu * mu) / (float)(NSEQ - 1);
      var = fmaxf(var, 0.f);
      stat[0] = mu;
      stat[1] = fmaxf(sqrtf(var), 1e-6f);
    }
    __syncthreads();
  }
  const float mean_e = stat[0];
  const float inv_std_e = 1.0f / stat[1];

  // 3 pooling levels (k=1,2,4; count_include_pad=False, crop st=0)
  for (int p = t; p < NSEQ; p += 256) {
    const int y = p >> 5, x = p & 31;
    // level 0 (k=1)
    float p0 = energy[p];
    // level 1 (k=2): rows [y-1,y], cols [x-1,x] clipped
    float s1 = 0.f; int c1 = 0;
#pragma unroll
    for (int dy = -1; dy <= 0; ++dy)
#pragma unroll
      for (int dx = -1; dx <= 0; ++dx) {
        int yy = y + dy, xx = x + dx;
        if (yy >= 0 && yy < 32 && xx >= 0 && xx < 32) { s1 += energy[yy * 32 + xx]; ++c1; }
      }
    float p1 = s1 / (float)c1;
    // level 2 (k=4): rows [y-2,y+1], cols [x-2,x+1] clipped
    float s2 = 0.f; int c2 = 0;
#pragma unroll
    for (int dy = -2; dy <= 1; ++dy)
#pragma unroll
      for (int dx = -2; dx <= 1; ++dx) {
        int yy = y + dy, xx = x + dx;
        if (yy >= 0 && yy < 32 && xx >= 0 && xx < 32) { s2 += energy[yy * 32 + xx]; ++c2; }
      }
    float p2 = s2 / (float)c2;

    float g0 = 1.0f / (1.0f + __expf(-(p0 - mean_e) * inv_std_e));
    float g1 = 1.0f / (1.0f + __expf(-(p1 - mean_e) * inv_std_e));
    float g2 = 1.0f / (1.0f + __expf(-(p2 - mean_e) * inv_std_e));
    frac[p] = (g0 + g1 + g2) * (1.0f / 3.0f);
  }
  __syncthreads();

  // standardize frac (ddof=1) -> ns
  {
    float s1 = 0.f, s2 = 0.f;
    for (int i = t; i < NSEQ; i += 256) { float e = frac[i]; s1 += e; s2 += e * e; }
    s1 = wave_sum(s1); s2 = wave_sum(s2);
    if (lane == 0) { rbuf[0][wave] = s1; rbuf[1][wave] = s2; }
    __syncthreads();
    if (t == 0) {
      float a = 0.f, c = 0.f;
      for (int i = 0; i < 8; ++i) { a += rbuf[0][i]; c += rbuf[1][i]; }
      float mu = a / (float)NSEQ;
      float var = (c - (float)NSEQ * mu * mu) / (float)(NSEQ - 1);
      var = fmaxf(var, 0.f);
      stat[0] = mu;
      stat[1] = fmaxf(sqrtf(var), 1e-6f);
    }
    __syncthreads();
  }
  const float mu_f = stat[0];
  const float inv_std_f = 1.0f / stat[1];
  for (int p = t; p < NSEQ; p += 256) ns[b * NSEQ + p] = (frac[p] - mu_f) * inv_std_f;
}

// ---------------------------------------------------------------------------
// Kernel: Y = X @ W^T + bias, f16 in, f16 out. Block: 256 thr (8 waves),
// tile 128(M) x 64(N); each wave = 16 rows x 64 cols (4 WMMA accumulators).
// transposeV: store per-batch transposed [D, NSEQ] for the V operand.
// ---------------------------------------------------------------------------
__global__ __launch_bounds__(256)
void gemm_qkv_kernel(const _Float16* __restrict__ X, const _Float16* __restrict__ W,
                     const float* __restrict__ bias, _Float16* __restrict__ out,
                     int transposeV) {
  const int lane = threadIdx.x & 31;
  const int wave = threadIdx.x >> 5;
  const int r0 = blockIdx.x * 128 + wave * 16;
  const int c0 = blockIdx.y * 64;

  v8f acc[4] = {v8f{}, v8f{}, v8f{}, v8f{}};
  const _Float16* xrow = X + (size_t)r0 * D_DIM;
  for (int k0 = 0; k0 < D_DIM; k0 += 32) {
    __builtin_prefetch((const void*)(xrow + k0 + 64), 0, 1);  // global_prefetch_b8
    v16h a = load_a(xrow + k0, D_DIM, lane);
#pragma unroll
    for (int tI = 0; tI < 4; ++tI) {
      v16h bf = load_b(W + (size_t)(c0 + tI * 16) * D_DIM + k0, D_DIM, lane);
      acc[tI] = wmma_f16(a, bf, acc[tI]);
    }
  }

  const int idx = lane & 15, hi = lane >> 4;
#pragma unroll
  for (int tI = 0; tI < 4; ++tI) {
    const int col = c0 + tI * 16 + idx;
    const float bv = bias[col];
    if (!transposeV) {
#pragma unroll
      for (int r = 0; r < 8; ++r) {
        int row = r0 + r + 8 * hi;
        out[(size_t)row * D_DIM + col] = (_Float16)(acc[tI][r] + bv);
      }
    } else {
      // per-batch transposed: Vt[b][j][n], b = row/1024, n = row%1024
      const int bidx = r0 >> 10;
      const int nbase = (r0 & (NSEQ - 1)) + 8 * hi;
      v8h pack;
#pragma unroll
      for (int r = 0; r < 8; ++r) pack[r] = (_Float16)(acc[tI][r] + bv);
      *(v8h*)(out + (size_t)bidx * D_DIM * NSEQ + (size_t)col * NSEQ + nbase) = pack;
    }
  }
}

// ---------------------------------------------------------------------------
// Kernel: fused attention for one 16-row query block of one batch.
// Grid (64, 16); 256 threads (8 waves). LDS: 16x1024 f32 logits/probs panel.
// ---------------------------------------------------------------------------
__global__ __launch_bounds__(256)
void attn_kernel(const _Float16* __restrict__ Qh, const _Float16* __restrict__ Kh,
                 const _Float16* __restrict__ Vt, const float* __restrict__ ns,
                 _Float16* __restrict__ Oh) {
  __shared__ float S[16 * NSEQ];   // 64 KB

  const int b = blockIdx.y;
  const int q0 = blockIdx.x * 16;
  const int lane = threadIdx.x & 31;
  const int wave = threadIdx.x >> 5;
  const int idx = lane & 15, hi = lane >> 4;

  const _Float16* Qb = Qh + ((size_t)b * NSEQ + q0) * D_DIM;
  const _Float16* Kb = Kh + (size_t)b * NSEQ * D_DIM;
  const _Float16* Vb = Vt + (size_t)b * D_DIM * NSEQ;
  const float* nsb = ns + b * NSEQ;

  // ---- phase 1: logits + bias -> LDS. wave handles 128 key columns. ----
  {
    const int cw = wave * 128;
    v8f acc[8] = {v8f{}, v8f{}, v8f{}, v8f{}, v8f{}, v8f{}, v8f{}, v8f{}};
    for (int k0 = 0; k0 < D_DIM; k0 += 32) {
      v16h a = load_a(Qb + k0, D_DIM, lane);
#pragma unroll
      for (int tI = 0; tI < 8; ++tI) {
        v16h bf = load_b(Kb + (size_t)(cw + tI * 16) * D_DIM + k0, D_DIM, lane);
        acc[tI] = wmma_f16(a, bf, acc[tI]);
      }
    }
    const float scale = 0.036084391824351615f;  // 1/sqrt(768)
    float nsq[8];
#pragma unroll
    for (int r = 0; r < 8; ++r) nsq[r] = nsb[q0 + r + 8 * hi];
#pragma unroll
    for (int tI = 0; tI < 8; ++tI) {
      const int c = cw + tI * 16 + idx;
      const float nc = nsb[c];
#pragma unroll
      for (int r = 0; r < 8; ++r)
        S[(r + 8 * hi) * NSEQ + c] = acc[tI][r] * scale - 2.0f * fabsf(nsq[r] - nc);
    }
  }
  __syncthreads();

  // ---- phase 2: exact softmax, one wave per 2 rows, wave32 reductions ----
  for (int row = wave * 2; row < wave * 2 + 2; ++row) {
    float* Sr = &S[row * NSEQ];
    float m = -3.0e38f;
    for (int c = lane; c < NSEQ; c += 32) m = fmaxf(m, Sr[c]);
    m = wave_max(m);
    float s = 0.f;
    for (int c = lane; c < NSEQ; c += 32) s += __expf(Sr[c] - m);
    s = wave_sum(s);
    const float inv = 1.0f / s;
    for (int c = lane; c < NSEQ; c += 32) Sr[c] = __expf(Sr[c] - m) * inv;
  }
  __syncthreads();

  // ---- phase 3: O = P @ V. wave handles 96 output dims (6 tiles). ----
  {
    const int j0 = wave * 96;
    v8f oacc[6] = {v8f{}, v8f{}, v8f{}, v8f{}, v8f{}, v8f{}};
    for (int k0 = 0; k0 < NSEQ; k0 += 32) {
      v16h a = load_a_f32(&S[k0], NSEQ, lane);
#pragma unroll
      for (int tI = 0; tI < 6; ++tI) {
        v16h bf = load_b(Vb + (size_t)(j0 + tI * 16) * NSEQ + k0, NSEQ, lane);
        oacc[tI] = wmma_f16(a, bf, oacc[tI]);
      }
    }
#pragma unroll
    for (int tI = 0; tI < 6; ++tI) {
      const int j = j0 + tI * 16 + idx;
#pragma unroll
      for (int r = 0; r < 8; ++r)
        Oh[((size_t)b * NSEQ + q0 + r + 8 * hi) * D_DIM + j] = (_Float16)oacc[tI][r];
    }
  }
}

// ---------------------------------------------------------------------------
// Kernel: out = feats + O @ Wo^T + bo   (f32 output)
// ---------------------------------------------------------------------------
__global__ __launch_bounds__(256)
void out_proj_kernel(const _Float16* __restrict__ Oh, const _Float16* __restrict__ Woh,
                     const float* __restrict__ bo, const float* __restrict__ feats,
                     float* __restrict__ out) {
  const int lane = threadIdx.x & 31;
  const int wave = threadIdx.x >> 5;
  const int r0 = blockIdx.x * 128 + wave * 16;
  const int c0 = blockIdx.y * 64;

  v8f acc[4] = {v8f{}, v8f{}, v8f{}, v8f{}};
  const _Float16* orow = Oh + (size_t)r0 * D_DIM;
  for (int k0 = 0; k0 < D_DIM; k0 += 32) {
    v16h a = load_a(orow + k0, D_DIM, lane);
#pragma unroll
    for (int tI = 0; tI < 4; ++tI) {
      v16h bf = load_b(Woh + (size_t)(c0 + tI * 16) * D_DIM + k0, D_DIM, lane);
      acc[tI] = wmma_f16(a, bf, acc[tI]);
    }
  }
  const int idx = lane & 15, hi = lane >> 4;
#pragma unroll
  for (int tI = 0; tI < 4; ++tI) {
    const int col = c0 + tI * 16 + idx;
    const float bv = bo[col];
#pragma unroll
    for (int r = 0; r < 8; ++r) {
      const size_t off = (size_t)(r0 + r + 8 * hi) * D_DIM + col;
      out[off] = feats[off] + acc[tI][r] + bv;
    }
  }
}

// ---------------------------------------------------------------------------
// Host launcher
// ---------------------------------------------------------------------------
extern "C" void kernel_launch(void* const* d_in, const int* in_sizes, int n_in,
                              void* d_out, int out_size, void* d_ws, size_t ws_size,
                              hipStream_t stream) {
  const float* feats = (const float*)d_in[0];
  const float* Wq = (const float*)d_in[1];
  const float* bq = (const float*)d_in[2];
  const float* Wk = (const float*)d_in[3];
  const float* bk = (const float*)d_in[4];
  const float* Wv = (const float*)d_in[5];
  const float* bv = (const float*)d_in[6];
  const float* Wo = (const float*)d_in[7];
  const float* bo = (const float*)d_in[8];
  float* out = (float*)d_out;

  // workspace layout
  char* ws = (char*)d_ws;
  float* ns = (float*)ws;                 ws += (size_t)MTOT * sizeof(float);
  _Float16* Xh  = (_Float16*)ws;          ws += (size_t)MTOT * D_DIM * 2;
  _Float16* Qh  = (_Float16*)ws;          ws += (size_t)MTOT * D_DIM * 2;
  _Float16* Kh  = (_Float16*)ws;          ws += (size_t)MTOT * D_DIM * 2;
  _Float16* Vt  = (_Float16*)ws;          ws += (size_t)MTOT * D_DIM * 2;
  _Float16* Oh  = (_Float16*)ws;          ws += (size_t)MTOT * D_DIM * 2;
  _Float16* Wqh = (_Float16*)ws;          ws += (size_t)D_DIM * D_DIM * 2;
  _Float16* Wkh = (_Float16*)ws;          ws += (size_t)D_DIM * D_DIM * 2;
  _Float16* Wvh = (_Float16*)ws;          ws += (size_t)D_DIM * D_DIM * 2;
  _Float16* Woh = (_Float16*)ws;

  const int nX = MTOT * D_DIM;
  const int nW = D_DIM * D_DIM;
  cvt_f32_to_f16<<<(nX + 255) / 256, 256, 0, stream>>>(feats, Xh, nX);
  cvt_f32_to_f16<<<(nW + 255) / 256, 256, 0, stream>>>(Wq, Wqh, nW);
  cvt_f32_to_f16<<<(nW + 255) / 256, 256, 0, stream>>>(Wk, Wkh, nW);
  cvt_f32_to_f16<<<(nW + 255) / 256, 256, 0, stream>>>(Wv, Wvh, nW);
  cvt_f32_to_f16<<<(nW + 255) / 256, 256, 0, stream>>>(Wo, Woh, nW);

  fractal_ns_kernel<<<NBATCH, 256, 0, stream>>>(feats, ns);

  dim3 gGemm(MTOT / 128, D_DIM / 64);   // (128, 12)
  gemm_qkv_kernel<<<gGemm, 256, 0, stream>>>(Xh, Wqh, bq, Qh, 0);
  gemm_qkv_kernel<<<gGemm, 256, 0, stream>>>(Xh, Wkh, bk, Kh, 0);
  gemm_qkv_kernel<<<gGemm, 256, 0, stream>>>(Xh, Wvh, bv, Vt, 1);

  attn_kernel<<<dim3(NSEQ / 16, NBATCH), 256, 0, stream>>>(Qh, Kh, Vt, ns, Oh);

  out_proj_kernel<<<gGemm, 256, 0, stream>>>(Oh, Woh, bo, feats, out);
}
